// OCTMamba_71021579207191
// MI455X (gfx1250) — matrix operand order repaired and checked by reading, not compiled, not run on hardware
//
#include <hip/hip_runtime.h>
#include <hip/hip_bf16.h>
#include <cstdint>

// ---------------------------------------------------------------------------
// OCTMamba forward for MI455X (gfx1250, wave32, WMMA).
// All channel contractions run through v_wmma_f32_16x16x32_f16:
//   - 3x3x3 convs as implicit GEMM (fused im2col gather on A, branch-free)
//   - mamba in_proj / x_proj / dt_proj / out_proj
//   - cfe 1x1x1 convs
// B fragments are pre-packed into the 32-lane WMMA register layout and staged
// block-cooperatively through LDS with global_load_async_to_lds_b128
// (double-buffered, s_wait_asynccnt + one barrier per K-chunk).
// ---------------------------------------------------------------------------

typedef __attribute__((ext_vector_type(16))) _Float16 v16h;
typedef __attribute__((ext_vector_type(8)))  float    v8f;
typedef __attribute__((ext_vector_type(8)))  unsigned v8u;

__device__ __forceinline__ float act_apply(float v, int act) {
    if (act == 1) return 0.5f * v * (1.0f + erff(v * 0.70710678118654752f)); // exact gelu
    if (act == 2) return v / (1.0f + expf(-v));                              // silu
    if (act == 3) return (v > 20.0f) ? v : log1pf(expf(v));                  // softplus
    return v;
}

__device__ __forceinline__ unsigned lds_off_of(const void* p) {
    // Low 32 bits of a generic pointer into LDS == LDS byte offset (aperture
    // high bits live in addr[63:32]).
    return (unsigned)(uintptr_t)p;
}

__device__ __forceinline__ void async_ldst_b128(unsigned ldsAddr, unsigned gOffBytes,
                                                const void* sbase) {
    asm volatile("global_load_async_to_lds_b128 %0, %1, %2 offset:0"
                 :: "v"(ldsAddr), "v"(gOffBytes), "s"(sbase) : "memory");
}

__device__ __forceinline__ void wait_async0() {
    asm volatile("s_wait_asynccnt 0x0" ::: "memory");
}

// ---------------------------------------------------------------------------
// Weight -> B-fragment packing.
// Logical B[k][n] = W[n*K + k]  (all torch-style (out, in...) weights).
// Fragment layout (per 16-col n-tile, per 32-deep k-chunk):
//   dword index = ((ntile*nKc + kc)*32 + lane)*8 + j
//   lane: n = ntile*16 + (lane&15), half = lane>>4
//   vgpr j holds f16 pair (k0,k0+1), k0 = kc*32 + half*16 + 2*j
// ---------------------------------------------------------------------------
__global__ void k_pack_b(const float* __restrict__ w, int K, int N,
                         int Kp, int Np, unsigned* __restrict__ frag) {
    int nKc = Kp >> 5;
    long long total = (long long)(Np >> 4) * nKc * 256;
    long long t = (long long)blockIdx.x * blockDim.x + threadIdx.x;
    if (t >= total) return;
    int j    = (int)(t & 7);
    int lane = (int)((t >> 3) & 31);
    long long rest = t >> 8;
    int kc = (int)(rest % nKc);
    int nt = (int)(rest / nKc);
    int n    = nt * 16 + (lane & 15);
    int half = lane >> 4;
    int k0   = kc * 32 + half * 16 + 2 * j;
    float f0 = 0.f, f1 = 0.f;
    if (n < N) {
        if (k0     < K) f0 = w[(long long)n * K + k0];
        if (k0 + 1 < K) f1 = w[(long long)n * K + k0 + 1];
    }
    union { _Float16 h[2]; unsigned u; } p;
    p.h[0] = (_Float16)f0; p.h[1] = (_Float16)f1;
    frag[(((long long)nt * nKc + kc) * 32 + lane) * 8 + j] = p.u;
}

// ---------------------------------------------------------------------------
// Generic WMMA GEMM:  C[m][n] = act( (sum_k A[m][k]*B[k][n]) [+bias][*scale+shift] ) [+resid]
// grid = (ceil(MT/8), NT): all 8 waves of a block share one n-tile, so B
// fragments are staged once per block per K-chunk via async LDS loads.
// ---------------------------------------------------------------------------
struct GemmP {
    const float* A;       long long aoff; int sM, sK;
    const unsigned* Bf;
    float* C;             long long coff; int cSM, cSN;
    const float* bias; const float* scale; const float* shift;
    const float* resid;   long long roff; int rSM, rSN;
    int M, N, K, Kp, act;
};

__global__ void __launch_bounds__(256) k_gemm_wmma(GemmP p) {
    __shared__ unsigned bsh[2][256];
    int tid  = threadIdx.x;
    int wv   = tid >> 5;
    int lane = tid & 31;
    int MT = (p.M + 15) >> 4;
    int mt = blockIdx.x * 8 + wv;
    int nt = blockIdx.y;
    bool active = (mt < MT);
    int half = lane >> 4, l15 = lane & 15;
    int mrow = mt * 16 + l15;
    bool mok = active && (mrow < p.M);
    int mclamp = (mrow < p.M) ? mrow : (p.M - 1);
    const float* Arow = p.A + p.aoff + (long long)mclamp * p.sM;

    int nKc = p.Kp >> 5;
    const unsigned* bfBase = p.Bf + (size_t)nt * nKc * 256;

    // stage chunk 0 (64 lanes x b128 = 1KB)
    if (tid < 64)
        async_ldst_b128(lds_off_of(&bsh[0][0]) + (unsigned)tid * 16u,
                        (unsigned)tid * 16u, bfBase);

    v8f acc = {};
    for (int kc = 0; kc < nKc; ++kc) {
        wait_async0();
        __syncthreads();
        if (kc + 1 < nKc && tid < 64)
            async_ldst_b128(lds_off_of(&bsh[(kc + 1) & 1][0]) + (unsigned)tid * 16u,
                            (unsigned)(kc + 1) * 1024u + (unsigned)tid * 16u, bfBase);

        union { v16h h; _Float16 e[16]; } a;
        int kbase = kc * 32;
#pragma unroll
        for (int j = 0; j < 8; ++j) {
            int kk = half * 16 + j * 2;
            int k0 = kbase + kk;
            int k1 = k0 + 1;
            int k0c = (k0 < p.K) ? k0 : (p.K - 1);
            int k1c = (k1 < p.K) ? k1 : (p.K - 1);
            float f0 = Arow[(long long)k0c * p.sK];
            float f1 = Arow[(long long)k1c * p.sK];
            f0 = (mok && k0 < p.K) ? f0 : 0.f;
            f1 = (mok && k1 < p.K) ? f1 : 0.f;
            a.e[2 * j]     = (_Float16)f0;
            a.e[2 * j + 1] = (_Float16)f1;
        }
        union { v8u u; v16h h; } b;
        b.u = *(const v8u*)(&bsh[kc & 1][lane * 8]);
        acc = __builtin_amdgcn_wmma_f32_16x16x32_f16(false, a.h, false, b.h,
                                                     (short)0, acc, false, false);
    }
    if (active) {
#pragma unroll
        for (int r = 0; r < 8; ++r) {
            int m = mt * 16 + half * 8 + r;
            int n = nt * 16 + l15;
            if (m < p.M && n < p.N) {
                float v = acc[r];
                if (p.bias)  v += p.bias[n];
                if (p.scale) v = v * p.scale[n] + (p.shift ? p.shift[n] : 0.f);
                v = act_apply(v, p.act);
                if (p.resid) v += p.resid[p.roff + (long long)m * p.rSM + (long long)n * p.rSN];
                p.C[p.coff + (long long)m * p.cSM + (long long)n * p.cSN] = v;
            }
        }
    }
}

// ---------------------------------------------------------------------------
// Implicit-GEMM 3x3x3 conv (pad=1): M = voxels, N = Cout, K = Cin*27.
// Branch-free fused im2col gather; B staged via async LDS like k_gemm_wmma.
// Output channel-major [Cout, V].
// ---------------------------------------------------------------------------
struct ConvP {
    const float* x; int Cin, Dd, Hh, Ww;
    const unsigned* Bf; int Kp;
    float* C;
    const float* bias;
    int Cout, act;
};

__global__ void __launch_bounds__(256) k_conv3_wmma(ConvP p) {
    __shared__ unsigned bsh[2][256];
    int tid  = threadIdx.x;
    int wv   = tid >> 5;
    int lane = tid & 31;
    int V = p.Dd * p.Hh * p.Ww;
    int K = p.Cin * 27;
    int MT = (V + 15) >> 4;
    int mt = blockIdx.x * 8 + wv;
    int nt = blockIdx.y;
    bool active = (mt < MT);
    int half = lane >> 4, l15 = lane & 15;
    int mrow = mt * 16 + l15;
    bool mok = active && (mrow < V);
    int mclamp = (mrow < V) ? mrow : (V - 1);
    int vd = mclamp / (p.Hh * p.Ww);
    int vh = (mclamp / p.Ww) % p.Hh;
    int vw = mclamp % p.Ww;

    int nKc = p.Kp >> 5;
    const unsigned* bfBase = p.Bf + (size_t)nt * nKc * 256;

    if (tid < 64)
        async_ldst_b128(lds_off_of(&bsh[0][0]) + (unsigned)tid * 16u,
                        (unsigned)tid * 16u, bfBase);

    v8f acc = {};
    for (int kc = 0; kc < nKc; ++kc) {
        wait_async0();
        __syncthreads();
        if (kc + 1 < nKc && tid < 64)
            async_ldst_b128(lds_off_of(&bsh[(kc + 1) & 1][0]) + (unsigned)tid * 16u,
                            (unsigned)(kc + 1) * 1024u + (unsigned)tid * 16u, bfBase);

        union { v16h h; _Float16 e[16]; } a;
        int kbase = kc * 32;
#pragma unroll
        for (int j = 0; j < 8; ++j) {
            int kk = half * 16 + j * 2;
#pragma unroll
            for (int q = 0; q < 2; ++q) {
                int k0 = kbase + kk + q;
                int kcl = (k0 < K) ? k0 : (K - 1);
                int ci = kcl / 27;
                int r  = kcl % 27;
                int zi = vd + r / 9 - 1;
                int yi = vh + (r / 3) % 3 - 1;
                int xi = vw + r % 3 - 1;
                bool inb = mok && (k0 < K) &&
                           (zi >= 0) && (zi < p.Dd) &&
                           (yi >= 0) && (yi < p.Hh) &&
                           (xi >= 0) && (xi < p.Ww);
                int zc = (zi < 0) ? 0 : ((zi >= p.Dd) ? p.Dd - 1 : zi);
                int yc = (yi < 0) ? 0 : ((yi >= p.Hh) ? p.Hh - 1 : yi);
                int xc = (xi < 0) ? 0 : ((xi >= p.Ww) ? p.Ww - 1 : xi);
                float f = p.x[(long long)ci * V + ((long long)zc * p.Hh + yc) * p.Ww + xc];
                a.e[2 * j + q] = (_Float16)(inb ? f : 0.f);
            }
        }
        union { v8u u; v16h h; } b;
        b.u = *(const v8u*)(&bsh[kc & 1][lane * 8]);
        acc = __builtin_amdgcn_wmma_f32_16x16x32_f16(false, a.h, false, b.h,
                                                     (short)0, acc, false, false);
    }
    if (active) {
#pragma unroll
        for (int r = 0; r < 8; ++r) {
            int m = mt * 16 + half * 8 + r;
            int n = nt * 16 + l15;
            if (m < V && n < p.Cout) {
                float v = acc[r];
                if (p.bias) v += p.bias[n];
                p.C[(long long)n * V + m] = act_apply(v, p.act);
            }
        }
    }
}

// ---------------------------------------------------------------------------
// Elementwise / reduction kernels
// ---------------------------------------------------------------------------

// Channel / group norm. One block per group (cpg contiguous channels of V).
__global__ void k_chan_norm(const float* __restrict__ x, float* __restrict__ o,
                            const float* __restrict__ g, const float* __restrict__ b,
                            int V, int cpg, int act) {
    __shared__ float ss[256], sq[256];
    long long base = (long long)blockIdx.x * cpg * V;
    int n = cpg * V;
    float s = 0.f, q = 0.f;
    for (int i = threadIdx.x; i < n; i += blockDim.x) {
        float v = x[base + i]; s += v; q += v * v;
    }
    ss[threadIdx.x] = s; sq[threadIdx.x] = q;
    __syncthreads();
    for (int st = 128; st > 0; st >>= 1) {
        if ((int)threadIdx.x < st) { ss[threadIdx.x] += ss[threadIdx.x + st];
                                     sq[threadIdx.x] += sq[threadIdx.x + st]; }
        __syncthreads();
    }
    float m  = ss[0] / n;
    float vr = sq[0] / n - m * m;
    float rs = rsqrtf(vr + 1e-5f);
    for (int i = threadIdx.x; i < n; i += blockDim.x) {
        int c = blockIdx.x * cpg + i / V;
        float v = (x[base + i] - m) * rs * g[c] + b[c];
        o[base + i] = act_apply(v, act);
    }
}

// Per-token layernorm, [C,V] strided in -> [V,C] token-major out.
__global__ void k_layernorm(const float* __restrict__ x, float* __restrict__ o,
                            const float* __restrict__ g, const float* __restrict__ b,
                            int C, int V) {
    __shared__ float ss[256], sq[256];
    int v = blockIdx.x;
    float s = 0.f, q = 0.f;
    for (int c = threadIdx.x; c < C; c += blockDim.x) {
        float t = x[(long long)c * V + v]; s += t; q += t * t;
    }
    ss[threadIdx.x] = s; sq[threadIdx.x] = q;
    __syncthreads();
    for (int st = 128; st > 0; st >>= 1) {
        if ((int)threadIdx.x < st) { ss[threadIdx.x] += ss[threadIdx.x + st];
                                     sq[threadIdx.x] += sq[threadIdx.x + st]; }
        __syncthreads();
    }
    float m  = ss[0] / C;
    float vr = sq[0] / C - m * m;
    float rs = rsqrtf(vr + 1e-5f);
    for (int c = threadIdx.x; c < C; c += blockDim.x) {
        float t = (x[(long long)c * V + v] - m) * rs * g[c] + b[c];
        o[(long long)v * C + c] = t;
    }
}

// BN affine coefficients: scale = g*rsqrt(rv+eps), shift = b - rm*scale.
__global__ void k_bn_coeffs(const float* g, const float* b, const float* rm,
                            const float* rv, float* sc, float* sh, int C) {
    int i = blockIdx.x * blockDim.x + threadIdx.x;
    if (i >= C) return;
    float s = g[i] * rsqrtf(rv[i] + 1e-5f);
    sc[i] = s; sh[i] = b[i] - rm[i] * s;
}

// Depthwise 3x3x3 conv, pad=1, no bias.
__global__ void k_dwconv3(const float* __restrict__ x, const float* __restrict__ w,
                          float* __restrict__ o, int C, int Dd, int Hh, int Ww) {
    long long V = (long long)Dd * Hh * Ww;
    long long idx = (long long)blockIdx.x * blockDim.x + threadIdx.x;
    if (idx >= (long long)C * V) return;
    int c = (int)(idx / V);
    int v = (int)(idx % V);
    int d = v / (Hh * Ww), h = (v / Ww) % Hh, wv = v % Ww;
    const float* wc = w + (long long)c * 27;
    const float* xc = x + (long long)c * V;
    float s = 0.f;
#pragma unroll
    for (int kd = 0; kd < 3; ++kd)
#pragma unroll
        for (int kh = 0; kh < 3; ++kh)
#pragma unroll
            for (int kw = 0; kw < 3; ++kw) {
                int zi = d + kd - 1, yi = h + kh - 1, xi = wv + kw - 1;
                bool inb = (zi >= 0 && zi < Dd && yi >= 0 && yi < Hh && xi >= 0 && xi < Ww);
                int zc = (zi < 0) ? 0 : ((zi >= Dd) ? Dd - 1 : zi);
                int yc = (yi < 0) ? 0 : ((yi >= Hh) ? Hh - 1 : yi);
                int xcv = (xi < 0) ? 0 : ((xi >= Ww) ? Ww - 1 : xi);
                float t = xc[((long long)zc * Hh + yc) * Ww + xcv];
                s += wc[kd * 9 + kh * 3 + kw] * (inb ? t : 0.f);
            }
    o[idx] = s;
}

// SFE combine: bn(dw) + bn(x * pw_w) + x
__global__ void k_sfe_combine(const float* __restrict__ dw, const float* __restrict__ x,
                              const float* __restrict__ pw,
                              const float* s1, const float* t1,
                              const float* s2, const float* t2,
                              float* __restrict__ o, int C, int V) {
    long long idx = (long long)blockIdx.x * blockDim.x + threadIdx.x;
    if (idx >= (long long)C * V) return;
    int c = (int)(idx / V);
    float xv = x[idx];
    o[idx] = (dw[idx] * s1[c] + t1[c]) + (xv * pw[c] * s2[c] + t2[c]) + xv;
}

// 2x2x2 max pool
__global__ void k_maxpool(const float* __restrict__ x, float* __restrict__ o,
                          int C, int Dd, int Hh, int Ww) {
    int oD = Dd >> 1, oH = Hh >> 1, oW = Ww >> 1;
    long long oV = (long long)oD * oH * oW;
    long long idx = (long long)blockIdx.x * blockDim.x + threadIdx.x;
    if (idx >= (long long)C * oV) return;
    int c = (int)(idx / oV);
    int v = (int)(idx % oV);
    int d = v / (oH * oW), h = (v / oW) % oH, wv = v % oW;
    const float* xc = x + (long long)c * Dd * Hh * Ww;
    float m = -3.402823466e+38f;
#pragma unroll
    for (int a = 0; a < 2; ++a)
#pragma unroll
        for (int bb = 0; bb < 2; ++bb)
#pragma unroll
            for (int cc = 0; cc < 2; ++cc) {
                float t = xc[((long long)(2 * d + a) * Hh + (2 * h + bb)) * Ww + (2 * wv + cc)];
                m = fmaxf(m, t);
            }
    o[idx] = m;
}

// out = concat(up2(a), b [, c]) at full-res DxHxW, channel-major.
__global__ void k_up2cat(const float* __restrict__ a, int Ca, int aD, int aH, int aW,
                         const float* __restrict__ b, int Cb,
                         const float* __restrict__ c, int Cc,
                         float* __restrict__ o, int Dd, int Hh, int Ww) {
    long long V = (long long)Dd * Hh * Ww;
    long long total = (long long)(Ca + Cb + Cc) * V;
    long long idx = (long long)blockIdx.x * blockDim.x + threadIdx.x;
    if (idx >= total) return;
    int ch = (int)(idx / V);
    int v  = (int)(idx % V);
    int d = v / (Hh * Ww), h = (v / Ww) % Hh, wv = v % Ww;
    float val;
    if (ch < Ca) {
        val = a[(long long)ch * aD * aH * aW + ((long long)(d >> 1) * aH + (h >> 1)) * aW + (wv >> 1)];
    } else if (ch < Ca + Cb) {
        val = b[(long long)(ch - Ca) * V + v];
    } else {
        val = c[(long long)(ch - Ca - Cb) * V + v];
    }
    o[idx] = val;
}

// Mamba causal depthwise conv1d (k=4, left pad 3) + bias + silu.
// Input columns [0,di) of xz rows (stride twodi). Output dense [L,di].
__global__ void k_mconv1d(const float* __restrict__ xz, int twodi,
                          const float* __restrict__ w, const float* __restrict__ b,
                          float* __restrict__ xc, int L, int di) {
    long long idx = (long long)blockIdx.x * blockDim.x + threadIdx.x;
    if (idx >= (long long)L * di) return;
    int l = (int)(idx / di);
    int c = (int)(idx % di);
    float s = 0.f;
#pragma unroll
    for (int j = 0; j < 4; ++j) {
        int ll = l - 3 + j;
        int lc = (ll < 0) ? 0 : ll;
        float t = xz[(long long)lc * twodi + c];
        s += w[c * 4 + j] * ((ll >= 0) ? t : 0.f);
    }
    s += b[c];
    xc[idx] = s / (1.0f + expf(-s));
}

// Selective-scan: one thread per inner channel, sequential over L, 16 states.
__global__ void k_scan(const float* __restrict__ dt, const float* __restrict__ xc,
                       const float* __restrict__ xdbl, int dr,
                       const float* __restrict__ A_log, const float* __restrict__ Dp,
                       const float* __restrict__ xz, int twodi,
                       float* __restrict__ y, int L, int di) {
    int c = blockIdx.x * blockDim.x + threadIdx.x;
    if (c >= di) return;
    float Ar[16];
#pragma unroll
    for (int s = 0; s < 16; ++s) Ar[s] = -expf(A_log[(long long)c * 16 + s]);
    float h[16];
#pragma unroll
    for (int s = 0; s < 16; ++s) h[s] = 0.f;
    float Dc = Dp[c];
    int rw = dr + 32;
    for (int l = 0; l < L; ++l) {
        float dtv = dt[(long long)l * di + c];
        float xv  = xc[(long long)l * di + c];
        const float* row = xdbl + (long long)l * rw;
        float acc = 0.f;
#pragma unroll
        for (int s = 0; s < 16; ++s) {
            float Bv = row[dr + s];
            float Cv = row[dr + 16 + s];
            h[s] = h[s] * expf(dtv * Ar[s]) + dtv * Bv * xv;
            acc += h[s] * Cv;
        }
        acc += Dc * xv;
        float zv = xz[(long long)l * twodi + di + c];
        acc *= zv / (1.0f + expf(-zv));
        y[(long long)l * di + c] = acc;
    }
}

// Final 1x1x1 seg conv: 16 -> 1.
__global__ void k_seg(const float* __restrict__ x, const float* __restrict__ w,
                      const float* __restrict__ b, float* __restrict__ o,
                      int C, int V) {
    int v = blockIdx.x * blockDim.x + threadIdx.x;
    if (v >= V) return;
    float s = b[0];
    for (int c = 0; c < C; ++c) s += w[c] * x[(long long)c * V + v];
    o[v] = s;
}

// ---------------------------------------------------------------------------
// Host orchestration
// ---------------------------------------------------------------------------

struct WS {
    char* base; size_t off, cap;
    void* alloc(size_t bytes) {
        size_t a = (off + 255) & ~(size_t)255;
        off = a + bytes;
        return (void*)(base + a);
    }
    float*    f(size_t n) { return (float*)alloc(n * sizeof(float)); }
    unsigned* u(size_t n) { return (unsigned*)alloc(n * sizeof(unsigned)); }
};

struct BnPms   { const float *g, *b, *rm, *rv; };
struct CnPms   { const float* w; BnPms bn; };
struct MambaPm { const float *in_proj, *conv1d_w, *conv1d_b, *x_proj,
                              *dt_proj_w, *dt_proj_b, *A_log, *D, *out_proj; };
struct BlockPm { const float *conv_w, *conv_b, *gn_g, *gn_b;
                 CnPms dw, pw; const float *ln_g, *ln_b;
                 MambaPm m; CnPms cfe1, cfe2; };
struct FePm    { const float *w, *b, *g, *bb; };

static inline int ceil16(int x) { return (x + 15) & ~15; }
static inline int ceil32(int x) { return (x + 31) & ~31; }

static const unsigned* packB(const float* w, int K, int N, int& Kp,
                             WS& ws, hipStream_t s) {
    Kp = ceil32(K);
    int Np = ceil16(N);
    size_t dwords = (size_t)(Np / 16) * (Kp / 32) * 256;
    unsigned* f = ws.u(dwords);
    unsigned blocks = (unsigned)((dwords + 255) / 256);
    k_pack_b<<<blocks, 256, 0, s>>>(w, K, N, Kp, Np, f);
    return f;
}

static void launch_gemm(const GemmP& p, hipStream_t s) {
    int MT = (p.M + 15) / 16, NT = (p.N + 15) / 16;
    dim3 grid((unsigned)((MT + 7) / 8), (unsigned)NT);
    k_gemm_wmma<<<grid, 256, 0, s>>>(p);
}

static void launch_conv(const ConvP& p, hipStream_t s) {
    int V = p.Dd * p.Hh * p.Ww;
    int MT = (V + 15) / 16, NT = (p.Cout + 15) / 16;
    dim3 grid((unsigned)((MT + 7) / 8), (unsigned)NT);
    k_conv3_wmma<<<grid, 256, 0, s>>>(p);
}

static inline unsigned gridOf(long long n, int tpb) {
    return (unsigned)((n + tpb - 1) / tpb);
}

static void run_fe(const FePm& P, const float* xin, int Cin, float* xout,
                   int Dd, int Hh, int Ww, WS& ws, hipStream_t s) {
    size_t mk = ws.off;
    int V = Dd * Hh * Ww;
    float* t0 = ws.f((size_t)16 * V);
    int Kp;
    const unsigned* bf = packB(P.w, Cin * 27, 16, Kp, ws, s);
    ConvP cp{}; cp.x = xin; cp.Cin = Cin; cp.Dd = Dd; cp.Hh = Hh; cp.Ww = Ww;
    cp.Bf = bf; cp.Kp = Kp; cp.C = t0; cp.bias = P.b; cp.Cout = 16; cp.act = 0;
    launch_conv(cp, s);
    k_chan_norm<<<16, 256, 0, s>>>(t0, xout, P.g, P.bb, V, 1, /*gelu*/1);
    ws.off = mk;
}

static void run_oct(const BlockPm& P, const float* xin, int Cin, int co,
                    int Dd, int Hh, int Ww, float* xout, WS& ws, hipStream_t s) {
    size_t mk = ws.off;
    int V = Dd * Hh * Ww;

    // 1. conv3x3x3 (+bias)
    float* t0 = ws.f((size_t)co * V);
    {
        int Kp;
        const unsigned* bf = packB(P.conv_w, Cin * 27, co, Kp, ws, s);
        ConvP cp{}; cp.x = xin; cp.Cin = Cin; cp.Dd = Dd; cp.Hh = Hh; cp.Ww = Ww;
        cp.Bf = bf; cp.Kp = Kp; cp.C = t0; cp.bias = P.conv_b; cp.Cout = co; cp.act = 0;
        launch_conv(cp, s);
    }
    // 2. groupnorm(groups=co/2) + gelu  -> idt
    float* xg = ws.f((size_t)co * V);
    k_chan_norm<<<co / 2, 256, 0, s>>>(t0, xg, P.gn_g, P.gn_b, V, 2, /*gelu*/1);

    // 3. SFE: bn(dwconv) + bn(pwconv) + x
    float* s1 = ws.f(co); float* t1 = ws.f(co);
    float* s2 = ws.f(co); float* t2 = ws.f(co);
    k_bn_coeffs<<<gridOf(co, 64), 64, 0, s>>>(P.dw.bn.g, P.dw.bn.b, P.dw.bn.rm, P.dw.bn.rv, s1, t1, co);
    k_bn_coeffs<<<gridOf(co, 64), 64, 0, s>>>(P.pw.bn.g, P.pw.bn.b, P.pw.bn.rm, P.pw.bn.rv, s2, t2, co);
    float* td = ws.f((size_t)co * V);
    k_dwconv3<<<gridOf((long long)co * V, 256), 256, 0, s>>>(xg, P.dw.w, td, co, Dd, Hh, Ww);
    float* xs = ws.f((size_t)co * V);
    k_sfe_combine<<<gridOf((long long)co * V, 256), 256, 0, s>>>(td, xg, P.pw.w, s1, t1, s2, t2, xs, co, V);

    // 4. layernorm -> token-major [V, co]
    float* tok = ws.f((size_t)V * co);
    k_layernorm<<<(unsigned)V, 256, 0, s>>>(xs, tok, P.ln_g, P.ln_b, co, V);

    // 5. mamba
    int di = 2 * co, dr = (co + 15) / 16, L = V, dxw = dr + 32;
    float* xz = ws.f((size_t)L * 2 * di);
    {
        int Kp; const unsigned* bf = packB(P.m.in_proj, co, 2 * di, Kp, ws, s);
        GemmP g{}; g.A = tok; g.sM = co; g.sK = 1; g.Bf = bf; g.Kp = Kp;
        g.C = xz; g.cSM = 2 * di; g.cSN = 1; g.M = L; g.N = 2 * di; g.K = co; g.act = 0;
        launch_gemm(g, s);
    }
    float* xc = ws.f((size_t)L * di);
    k_mconv1d<<<gridOf((long long)L * di, 256), 256, 0, s>>>(xz, 2 * di, P.m.conv1d_w, P.m.conv1d_b, xc, L, di);
    float* xdbl = ws.f((size_t)L * dxw);
    {
        int Kp; const unsigned* bf = packB(P.m.x_proj, di, dxw, Kp, ws, s);
        GemmP g{}; g.A = xc; g.sM = di; g.sK = 1; g.Bf = bf; g.Kp = Kp;
        g.C = xdbl; g.cSM = dxw; g.cSN = 1; g.M = L; g.N = dxw; g.K = di; g.act = 0;
        launch_gemm(g, s);
    }
    float* dt = ws.f((size_t)L * di);
    {
        int Kp; const unsigned* bf = packB(P.m.dt_proj_w, dr, di, Kp, ws, s);
        GemmP g{}; g.A = xdbl; g.sM = dxw; g.sK = 1; g.Bf = bf; g.Kp = Kp;
        g.C = dt; g.cSM = di; g.cSN = 1; g.bias = P.m.dt_proj_b;
        g.M = L; g.N = di; g.K = dr; g.act = 3; // softplus
        launch_gemm(g, s);
    }
    float* y = ws.f((size_t)L * di);
    k_scan<<<gridOf(di, 256), 256, 0, s>>>(dt, xc, xdbl, dr, P.m.A_log, P.m.D, xz, 2 * di, y, L, di);
    float* xo = ws.f((size_t)L * co);
    {
        int Kp; const unsigned* bf = packB(P.m.out_proj, di, co, Kp, ws, s);
        GemmP g{}; g.A = y; g.sM = di; g.sK = 1; g.Bf = bf; g.Kp = Kp;
        g.C = xo; g.cSM = co; g.cSN = 1; g.M = L; g.N = co; g.K = di; g.act = 0;
        launch_gemm(g, s);
    }

    // 6. cfe1 (1x1x1 conv, bn, gelu) — output channel-major [2co, V]
    int c1 = co * 2;
    float* sc1 = ws.f(c1); float* sh1 = ws.f(c1);
    k_bn_coeffs<<<gridOf(c1, 64), 64, 0, s>>>(P.cfe1.bn.g, P.cfe1.bn.b, P.cfe1.bn.rm, P.cfe1.bn.rv, sc1, sh1, c1);
    float* f1 = ws.f((size_t)c1 * V);
    {
        int Kp; const unsigned* bf = packB(P.cfe1.w, co, c1, Kp, ws, s);
        GemmP g{}; g.A = xo; g.sM = co; g.sK = 1; g.Bf = bf; g.Kp = Kp;
        g.C = f1; g.cSM = 1; g.cSN = V; g.scale = sc1; g.shift = sh1;
        g.M = V; g.N = c1; g.K = co; g.act = 1; // gelu
        launch_gemm(g, s);
    }
    // 7. cfe2 (1x1x1 conv, bn) + residual idt -> xout [co, V]
    float* sc2 = ws.f(co); float* sh2 = ws.f(co);
    k_bn_coeffs<<<gridOf(co, 64), 64, 0, s>>>(P.cfe2.bn.g, P.cfe2.bn.b, P.cfe2.bn.rm, P.cfe2.bn.rv, sc2, sh2, co);
    {
        int Kp; const unsigned* bf = packB(P.cfe2.w, c1, co, Kp, ws, s);
        GemmP g{}; g.A = f1; g.sM = 1; g.sK = V; g.Bf = bf; g.Kp = Kp;
        g.C = xout; g.cSM = 1; g.cSN = V; g.scale = sc2; g.shift = sh2;
        g.resid = xg; g.rSM = 1; g.rSN = V;
        g.M = V; g.N = co; g.K = c1; g.act = 0;
        launch_gemm(g, s);
    }
    ws.off = mk;
}

extern "C" void kernel_launch(void* const* d_in, const int* in_sizes, int n_in,
                              void* d_out, int out_size, void* d_ws, size_t ws_size,
                              hipStream_t stream) {
    (void)in_sizes; (void)n_in; (void)out_size;
    WS ws{(char*)d_ws, 0, ws_size};

    const float* input = (const float*)d_in[0];
    const float* coord = (const float*)d_in[1];
    int pi = 2;
    auto nx     = [&]() { return (const float*)d_in[pi++]; };
    auto readBn = [&]() { BnPms b; b.g = nx(); b.b = nx(); b.rm = nx(); b.rv = nx(); return b; };
    auto readCn = [&]() { CnPms c; c.w = nx(); c.bn = readBn(); return c; };
    auto readM  = [&]() { MambaPm m; m.in_proj = nx(); m.conv1d_w = nx(); m.conv1d_b = nx();
                          m.x_proj = nx(); m.dt_proj_w = nx(); m.dt_proj_b = nx();
                          m.A_log = nx(); m.D = nx(); m.out_proj = nx(); return m; };
    auto readBk = [&]() { BlockPm b; b.conv_w = nx(); b.conv_b = nx(); b.gn_g = nx(); b.gn_b = nx();
                          b.dw = readCn(); b.pw = readCn(); b.ln_g = nx(); b.ln_b = nx();
                          b.m = readM(); b.cfe1 = readCn(); b.cfe2 = readCn(); return b; };

    FePm fe0{nx(), nx(), nx(), nx()};
    FePm fe1{nx(), nx(), nx(), nx()};
    BlockPm enc1 = readBk(), enc2 = readBk(), enc3 = readBk(), enc4 = readBk();
    BlockPm dec1 = readBk(), dec2 = readBk(), dec3 = readBk(), dec4 = readBk();
    const float* seg_w = nx();
    const float* seg_b = nx();

    const int V32 = 32 * 32 * 32, V16 = 16 * 16 * 16, V8 = 8 * 8 * 8, V4 = 64, V2 = 8;

    // Feature extractor
    float* xf0 = ws.f((size_t)16 * V32);
    run_fe(fe0, input, 1, xf0, 32, 32, 32, ws, stream);
    float* xf = ws.f((size_t)16 * V32);
    run_fe(fe1, xf0, 16, xf, 32, 32, 32, ws, stream);

    // Encoder
    float* p1 = ws.f((size_t)16 * V16);
    k_maxpool<<<gridOf((long long)16 * V16, 256), 256, 0, stream>>>(xf, p1, 16, 32, 32, 32);
    float* e1 = ws.f((size_t)32 * V16);
    run_oct(enc1, p1, 16, 32, 16, 16, 16, e1, ws, stream);

    float* p2 = ws.f((size_t)32 * V8);
    k_maxpool<<<gridOf((long long)32 * V8, 256), 256, 0, stream>>>(e1, p2, 32, 16, 16, 16);
    float* e2 = ws.f((size_t)64 * V8);
    run_oct(enc2, p2, 32, 64, 8, 8, 8, e2, ws, stream);

    float* p3 = ws.f((size_t)64 * V4);
    k_maxpool<<<gridOf((long long)64 * V4, 256), 256, 0, stream>>>(e2, p3, 64, 8, 8, 8);
    float* e3 = ws.f((size_t)128 * V4);
    run_oct(enc3, p3, 64, 128, 4, 4, 4, e3, ws, stream);

    float* p4 = ws.f((size_t)128 * V2);
    k_maxpool<<<gridOf((long long)128 * V2, 256), 256, 0, stream>>>(e3, p4, 128, 4, 4, 4);
    float* e4 = ws.f((size_t)256 * V2);
    run_oct(enc4, p4, 128, 256, 2, 2, 2, e4, ws, stream);

    // Decoder
    float* u1 = ws.f((size_t)384 * V4);
    k_up2cat<<<gridOf((long long)384 * V4, 256), 256, 0, stream>>>(
        e4, 256, 2, 2, 2, e3, 128, nullptr, 0, u1, 4, 4, 4);
    float* d1 = ws.f((size_t)128 * V4);
    run_oct(dec1, u1, 384, 128, 4, 4, 4, d1, ws, stream);

    float* u2 = ws.f((size_t)192 * V8);
    k_up2cat<<<gridOf((long long)192 * V8, 256), 256, 0, stream>>>(
        d1, 128, 4, 4, 4, e2, 64, nullptr, 0, u2, 8, 8, 8);
    float* d2 = ws.f((size_t)64 * V8);
    run_oct(dec2, u2, 192, 64, 8, 8, 8, d2, ws, stream);

    float* u3 = ws.f((size_t)96 * V16);
    k_up2cat<<<gridOf((long long)96 * V16, 256), 256, 0, stream>>>(
        d2, 64, 8, 8, 8, e1, 32, nullptr, 0, u3, 16, 16, 16);
    float* d3 = ws.f((size_t)32 * V16);
    run_oct(dec3, u3, 96, 32, 16, 16, 16, d3, ws, stream);

    float* u4 = ws.f((size_t)51 * V32);
    k_up2cat<<<gridOf((long long)51 * V32, 256), 256, 0, stream>>>(
        d3, 32, 16, 16, 16, xf, 16, coord, 3, u4, 32, 32, 32);
    float* d4 = ws.f((size_t)16 * V32);
    run_oct(dec4, u4, 51, 16, 32, 32, 32, d4, ws, stream);

    // Seg head
    k_seg<<<gridOf(V32, 256), 256, 0, stream>>>(d4, seg_w, seg_b, (float*)d_out, 16, V32);
}
